// AlphaAlt_10041633538279
// MI455X (gfx1250) — matrix operand order repaired and checked by reading, not compiled, not run on hardware
//
#include <hip/hip_runtime.h>
#include <hip/hip_bf16.h>
#include <math.h>

typedef __attribute__((ext_vector_type(16))) _Float16 v16h;
typedef __attribute__((ext_vector_type(8)))  _Float16 v8h;
typedef __attribute__((ext_vector_type(8)))  float    v8f;

#define BATCH   256
#define LATENT  256
#define DMODEL  10
#define SEQLEN  128
#define SIGX    0.1f
#define SIGZ    0.1f
#define NBZ     65536           // 256*256 floats, one state slab
#define VL      8
#define LDA     544             // uniform f16 activation row stride (mult of 32)

// ---------------------------------------------------------------- utilities

__global__ void init_state_k(const float* __restrict__ nz0,
                             const float* __restrict__ eps_x,
                             const float* __restrict__ eps_z,
                             float* __restrict__ out, float* __restrict__ buf,
                             float* __restrict__ ex, float* __restrict__ ez) {
    int i = blockIdx.x * blockDim.x + threadIdx.x;
    if (i >= NBZ) return;
    float z0 = SIGX * nz0[i];
    out[i] = z0;
    buf[i] = z0;
#pragma unroll
    for (int s = 1; s < VL; s++) buf[s * NBZ + i] = 0.0f;
    ex[i] = eps_x[i];
    ez[i] = eps_z[i];
}

__global__ void eps_update_k(float* __restrict__ ex, float* __restrict__ ez,
                             const float* __restrict__ nx, const float* __restrict__ nz) {
    int i = blockIdx.x * blockDim.x + threadIdx.x;
    if (i >= NBZ) return;
    ez[i] = (ez[i] + SIGZ * nz[i]) * 0.5f;
    ex[i] = (ex[i] + SIGX * nx[i]) * 0.5f;
}

// ------------------------------------------- weight swizzle + bias padding
// B-fragment order: dst[(((kt*tilesN + tn)*32 + lane)*16) + e] = W[kt*32 + 16*(e>>3) + 8*(lane>>4) + (e&7)][tn*16 + (lane&15)]

__global__ void repack_w_k(const float* __restrict__ src, _Float16* __restrict__ dst,
                           int Kact, int Nact, int Kpad, int Npad) {
    int idx = blockIdx.x * blockDim.x + threadIdx.x;
    if (idx >= Kpad * Npad) return;
    int e    = idx & 15;
    int lane = (idx >> 4) & 31;
    int t    = idx >> 9;
    int tilesN = Npad >> 4;
    int kt = t / tilesN, tn = t % tilesN;
    int k = kt * 32 + ((e >> 3) << 4) + ((lane >> 4) << 3) + (e & 7);
    int n = tn * 16 + (lane & 15);
    _Float16 v = (_Float16)0.0f;
    if (k < Kact && n < Nact) v = (_Float16)src[k * Nact + n];
    dst[idx] = v;
}

__global__ void biaspad_k(const float* __restrict__ src, float* __restrict__ dst,
                          int Nact, int Npad) {
    int i = blockIdx.x * blockDim.x + threadIdx.x;
    if (i < Npad) dst[i] = (i < Nact) ? src[i] : 0.0f;
}

// --------------------------------------------------- Gram + Jacobi + alpha

__global__ __launch_bounds__(256)
void gram_alpha_k(const float* __restrict__ buf,
                  const float* __restrict__ aw, const float* __restrict__ ab,
                  float* __restrict__ scal, int k) {
    float acc[36];
#pragma unroll
    for (int p = 0; p < 36; p++) acc[p] = 0.0f;
    int tid = threadIdx.x;
    for (int e = tid; e < NBZ; e += 256) {
        float v[VL];
#pragma unroll
        for (int s = 0; s < VL; s++) v[s] = buf[s * NBZ + e];
        int p = 0;
#pragma unroll
        for (int i = 0; i < VL; i++)
#pragma unroll
            for (int j = i; j < VL; j++, p++) acc[p] += v[i] * v[j];
    }
#pragma unroll
    for (int off = 16; off > 0; off >>= 1) {
#pragma unroll
        for (int p = 0; p < 36; p++) acc[p] += __shfl_down(acc[p], off);
    }
    __shared__ float red[8][36];
    int wv = tid >> 5, ln = tid & 31;
    if (ln == 0) {
#pragma unroll
        for (int p = 0; p < 36; p++) red[wv][p] = acc[p];
    }
    __syncthreads();
    if (tid == 0) {
        float G[VL][VL];
        int p = 0;
        for (int i = 0; i < VL; i++)
            for (int j = i; j < VL; j++, p++) {
                float s = 0.0f;
                for (int w = 0; w < 8; w++) s += red[w][p];
                s *= (1.0f / (float)NBZ);
                G[i][j] = s; G[j][i] = s;
            }
        int count = (k < VL) ? k : VL;
        bool valid[VL];
        for (int s = 0; s < VL; s++) valid[s] = false;
        for (int j = k - count; j < k; j++) valid[j & (VL - 1)] = true;
        float Km[VL][VL];
        float inv_cnt = 1.0f / (float)count;
        for (int i = 0; i < VL; i++)
            for (int j = 0; j < VL; j++)
                Km[i][j] = (valid[i] && valid[j])
                               ? (G[i][i] + G[j][j] - 2.0f * G[i][j]) * inv_cnt
                               : 0.0f;
        for (int sweep = 0; sweep < 12; sweep++) {
            for (int i = 0; i < VL - 1; i++)
                for (int j = i + 1; j < VL; j++) {
                    float apq = Km[i][j];
                    if (fabsf(apq) < 1e-24f) continue;
                    float app = Km[i][i], aqq = Km[j][j];
                    float theta = 0.5f * (aqq - app) / apq;
                    float t = ((theta >= 0.0f) ? 1.0f : -1.0f) /
                              (fabsf(theta) + sqrtf(theta * theta + 1.0f));
                    float c = 1.0f / sqrtf(t * t + 1.0f);
                    float sn = t * c;
                    for (int r = 0; r < VL; r++) {
                        float arp = Km[r][i], arq = Km[r][j];
                        Km[r][i] = c * arp - sn * arq;
                        Km[r][j] = sn * arp + c * arq;
                    }
                    for (int r = 0; r < VL; r++) {
                        float apr = Km[i][r], aqr = Km[j][r];
                        Km[i][r] = c * apr - sn * aqr;
                        Km[j][r] = sn * apr + c * aqr;
                    }
                }
        }
        float s2 = 0.0f;
        for (int i = 0; i < VL; i++) {
            float w = Km[i][i];
            if (w > 0.0f) s2 += w * w;
        }
        float vs = 1.0f / fmaxf(s2, 1e-12f);              // q=2: exp(log(s)/(1-q)) = 1/s
        float a = (1.0f / (1.0f + expf(-(vs * aw[0] + ab[0])))) *
                  (1.0f - SIGZ * SIGZ - 1e-4f);
        scal[0] = sqrtf(a);
        scal[1] = sqrtf(fmaxf(1.0f - a - SIGZ * SIGZ, 0.0f));
    }
}

// ------------------------------------------------------- packing (+embed), f16 out

__device__ __forceinline__ float ts_embed(int idx, int k) {
    int j = (idx < 5) ? idx : idx - 5;
    float freq = expf((float)j * (-9.210340371976184f / 4.0f));
    float ang = (float)k * freq;
    return (idx < 5) ? sinf(ang) : cosf(ang);
}

__global__ void pack2_k(const float* __restrict__ x, _Float16* __restrict__ out, int k) {
    int i = blockIdx.x * blockDim.x + threadIdx.x;           // BATCH * 288
    if (i >= BATCH * 288) return;
    int r = i / 288, c = i % 288;
    float v = 0.0f;
    if (c < LATENT)              v = x[r * LATENT + c];
    else if (c < LATENT + DMODEL) v = ts_embed(c - LATENT, k);
    out[r * LDA + c] = (_Float16)v;
}

__global__ void pack3_k(const float* __restrict__ x, const float* __restrict__ h,
                        _Float16* __restrict__ out, int k) {
    int i = blockIdx.x * blockDim.x + threadIdx.x;           // BATCH * 544
    if (i >= BATCH * 544) return;
    int r = i / 544, c = i % 544;
    float v = 0.0f;
    if (c < LATENT)                   v = x[r * LATENT + c];
    else if (c < 2 * LATENT)          v = h[r * LATENT + (c - LATENT)];
    else if (c < 2 * LATENT + DMODEL) v = ts_embed(c - 2 * LATENT, k);
    out[r * LDA + c] = (_Float16)v;
}

// ------------------------------------------------------------ WMMA GEMM
// A: f16 [256 x Kpad], row stride LDA. Wsw: f16 swizzled B-fragments.
// gelu=1: C -> f16 (stride LDA) after exact GELU.  gelu=0: C -> f32 (stride Npad).
// One wave computes two adjacent 16x16 N-tiles (shared A fragment).

__global__ __launch_bounds__(128)
void gemm_wmma_k(const _Float16* __restrict__ A, const _Float16* __restrict__ Wsw,
                 const float* __restrict__ bias, float* __restrict__ Cf,
                 _Float16* __restrict__ Ch, int Npad, int Kpad, int gelu) {
    const int tilesN = Npad >> 4;
    const int pairsN = tilesN >> 1;
    int wp = blockIdx.x * (blockDim.x >> 5) + (threadIdx.x >> 5);
    if (wp >= 16 * pairsN) return;                   // wave-uniform exit
    int tm  = wp / pairsN;
    int tn0 = (wp % pairsN) * 2;
    int lane = threadIdx.x & 31;
    int hi = lane >> 4, lr = lane & 15;
    const _Float16* Ar = A + (tm * 16 + lr) * LDA + 8 * hi;
    const v16h* Bp = (const v16h*)Wsw + (size_t)tn0 * 32 + lane;
    const int ktiles = Kpad >> 5;
    v8f acc0 = {}, acc1 = {};
    for (int kt = 0; kt < ktiles; ++kt) {
        int k0 = kt << 5;
        v8h a0 = *(const v8h*)(Ar + k0);             // K = k0+8hi .. +7   (b128)
        v8h a1 = *(const v8h*)(Ar + k0 + 16);        // K = k0+16+8hi ..+7 (b128)
        v16h a = __builtin_shufflevector(a0, a1, 0, 1, 2, 3, 4, 5, 6, 7,
                                                 8, 9, 10, 11, 12, 13, 14, 15);
        v16h b0 = Bp[(size_t)kt * tilesN * 32];      // tile tn0
        v16h b1 = Bp[(size_t)kt * tilesN * 32 + 32]; // tile tn0+1
        acc0 = __builtin_amdgcn_wmma_f32_16x16x32_f16(false, a, false, b0,
                                                      (short)0, acc0, false, false);
        acc1 = __builtin_amdgcn_wmma_f32_16x16x32_f16(false, a, false, b1,
                                                      (short)0, acc1, false, false);
    }
    int col0 = tn0 * 16 + lr, col1 = col0 + 16;
    float bv0 = bias[col0], bv1 = bias[col1];
    if (gelu) {
#pragma unroll
        for (int v = 0; v < 8; ++v) {
            int r = tm * 16 + v + 8 * hi;
            float x0 = acc0[v] + bv0;
            float x1 = acc1[v] + bv1;
            x0 = 0.5f * x0 * (1.0f + erff(x0 * 0.7071067811865475f));
            x1 = 0.5f * x1 * (1.0f + erff(x1 * 0.7071067811865475f));
            Ch[r * LDA + col0] = (_Float16)x0;
            Ch[r * LDA + col1] = (_Float16)x1;
        }
    } else {
#pragma unroll
        for (int v = 0; v < 8; ++v) {
            int r = tm * 16 + v + 8 * hi;
            Cf[r * Npad + col0] = acc0[v] + bv0;
            Cf[r * Npad + col1] = acc1[v] + bv1;
        }
    }
}

// ------------------------------------------------------------- epilogues

__global__ void xk_k(const float* __restrict__ g, const float* __restrict__ ex,
                     float* __restrict__ xk) {
    int i = blockIdx.x * blockDim.x + threadIdx.x;
    if (i >= NBZ) return;
    xk[i] = 0.9949874371066199f * g[i] + ex[i];      // sqrt(1 - 0.01)
}

__global__ void zk_k(const float* __restrict__ f, const float* __restrict__ zprev,
                     const float* __restrict__ ez, const float* __restrict__ scal,
                     float* __restrict__ out, float* __restrict__ bufslot) {
    int i = blockIdx.x * blockDim.x + threadIdx.x;
    if (i >= NBZ) return;
    float z = scal[0] * f[i] + scal[1] * zprev[i] + ez[i];
    out[i] = z;
    bufslot[i] = z;
}

// ---------------------------------------------------------------- driver

extern "C" void kernel_launch(void* const* d_in, const int* in_sizes, int n_in,
                              void* d_out, int out_size, void* d_ws, size_t ws_size,
                              hipStream_t stream) {
    const float* eps_x   = (const float*)d_in[2];
    const float* eps_z   = (const float*)d_in[3];
    const float* nz0     = (const float*)d_in[4];
    const float* noise_z = (const float*)d_in[5];
    const float* noise_x = (const float*)d_in[6];
    const float* aw      = (const float*)d_in[7];
    const float* ab      = (const float*)d_in[8];

    const float* Wf32[12]; const float* Bf32[12];
    for (int n = 0; n < 4; n++)
        for (int l = 0; l < 3; l++) {
            Wf32[n * 3 + l] = (const float*)d_in[9 + n * 6 + l * 2];
            Bf32[n * 3 + l] = (const float*)d_in[9 + n * 6 + l * 2 + 1];
        }
    static const int NA[12] = {266, 266, 256, 522, 522, 256, 266, 266, 256, 522, 522, 256};
    static const int KA[12] = {266, 266, 266, 522, 522, 522, 266, 266, 266, 522, 522, 522};
    static const int NP[12] = {288, 288, 256, 544, 544, 256, 288, 288, 256, 544, 544, 256};
    static const int KP[12] = {288, 288, 288, 544, 544, 544, 288, 288, 288, 544, 544, 544};

    float* out = (float*)d_out;

    // ---- workspace carve (float units; every chunk keeps 32B alignment) ----
    float* ws    = (float*)d_ws;
    float* buf   = ws;  ws += VL * NBZ;
    float* ex    = ws;  ws += NBZ;
    float* ez    = ws;  ws += NBZ;
    float* hEnc  = ws;  ws += NBZ;
    float* gOut  = ws;  ws += NBZ;
    float* xK    = ws;  ws += NBZ;
    float* fOut  = ws;  ws += NBZ;
    float* scal  = ws;  ws += 64;
    float* Bp[12];
    for (int l = 0; l < 12; l++) { Bp[l] = ws; ws += 576; }      // padded bias
    _Float16* hp = (_Float16*)ws;
    _Float16* pA  = hp;  hp += BATCH * LDA;                      // packed input (f16)
    _Float16* h1  = hp;  hp += BATCH * LDA;
    _Float16* h2  = hp;  hp += BATCH * LDA;
    _Float16* W16[12];
    for (int l = 0; l < 12; l++) { W16[l] = hp; hp += NP[l] * KP[l]; }

    // ---- one-time weight swizzle (B-fragment order) + bias padding ----
    for (int l = 0; l < 12; l++) {
        int n = NP[l] * KP[l];
        repack_w_k<<<(n + 255) / 256, 256, 0, stream>>>(Wf32[l], W16[l],
                                                        KA[l], NA[l], KP[l], NP[l]);
        biaspad_k<<<(NP[l] + 255) / 256, 256, 0, stream>>>(Bf32[l], Bp[l], NA[l], NP[l]);
    }
    init_state_k<<<NBZ / 256, 256, 0, stream>>>(nz0, eps_x, eps_z, out, buf, ex, ez);

    auto gemm = [&](const _Float16* A, int l, float* Cf, _Float16* Ch, int gelu) {
        int pairs = 16 * (NP[l] >> 5);               // 16 row tiles * N tile-pairs
        int blocks = (pairs + 3) / 4;                // 4 waves / block
        gemm_wmma_k<<<blocks, 128, 0, stream>>>(A, W16[l], Bp[l], Cf, Ch,
                                                NP[l], KP[l], gelu);
    };

    for (int k = 1; k < SEQLEN; k++) {
        const float* zprev = out + (size_t)(k - 1) * NBZ;
        eps_update_k<<<NBZ / 256, 256, 0, stream>>>(ex, ez,
                                                    noise_x + (size_t)(k - 1) * NBZ,
                                                    noise_z + (size_t)(k - 1) * NBZ);
        gram_alpha_k<<<1, 256, 0, stream>>>(buf, aw, ab, scal, k);

        // ---- g net ----
        pack2_k<<<(BATCH * 288 + 255) / 256, 256, 0, stream>>>(zprev, pA, k);
        gemm(pA, 0, nullptr, h1, 1);
        gemm(h1, 1, nullptr, h2, 1);
        gemm(h2, 2, hEnc, nullptr, 0);
        pack3_k<<<(BATCH * 544 + 255) / 256, 256, 0, stream>>>(zprev, hEnc, pA, k);
        gemm(pA, 3, nullptr, h1, 1);
        gemm(h1, 4, nullptr, h2, 1);
        gemm(h2, 5, gOut, nullptr, 0);
        xk_k<<<NBZ / 256, 256, 0, stream>>>(gOut, ex, xK);

        // ---- f net ----
        pack2_k<<<(BATCH * 288 + 255) / 256, 256, 0, stream>>>(xK, pA, k);
        gemm(pA, 6, nullptr, h1, 1);
        gemm(h1, 7, nullptr, h2, 1);
        gemm(h2, 8, hEnc, nullptr, 0);
        pack3_k<<<(BATCH * 544 + 255) / 256, 256, 0, stream>>>(xK, hEnc, pA, k);
        gemm(pA, 9, nullptr, h1, 1);
        gemm(h1, 10, nullptr, h2, 1);
        gemm(h2, 11, fOut, nullptr, 0);

        zk_k<<<NBZ / 256, 256, 0, stream>>>(fOut, zprev, ez, scal,
                                            out + (size_t)k * NBZ,
                                            buf + (size_t)(k & (VL - 1)) * NBZ);
    }
}
